// CausalSelfAttentionWithKVCache_7009386627758
// MI455X (gfx1250) — matrix operand order repaired and checked by reading, not compiled, not run on hardware
//
#include <hip/hip_runtime.h>
#include <hip/hip_bf16.h>

// Problem constants (from reference)
#define B_    2
#define T_    2048
#define EMBD_ 1024
#define NH_   16
#define DK_   64

typedef __attribute__((ext_vector_type(16))) _Float16 v16h;
typedef __attribute__((ext_vector_type(8)))  _Float16 v8h;   // 16 bytes
typedef __attribute__((ext_vector_type(4)))  _Float16 v4h;
typedef __attribute__((ext_vector_type(8)))  float    v8f;
typedef __attribute__((ext_vector_type(4)))  float    v4f;

__device__ __forceinline__ v8f wmma_f16(v16h a, v16h b, v8f c) {
  // (neg_a, A, neg_b, B, c_mod, C, reuse_a, reuse_b)
  return __builtin_amdgcn_wmma_f32_16x16x32_f16(false, a, false, b, (short)0, c, false, false);
}

__device__ __forceinline__ v16h cat16(v8h lo, v8h hi) {
  return __builtin_shufflevector(lo, hi, 0, 1, 2, 3, 4, 5, 6, 7, 8, 9, 10, 11, 12, 13, 14, 15);
}

// ---------------------------------------------------------------------------
// Elementwise f32 -> f16 convert (for x). Vectorized 4-wide.
// ---------------------------------------------------------------------------
__global__ void cvt_f16_kernel(const float* __restrict__ in,
                               _Float16* __restrict__ out, int n4) {
  const int i = blockIdx.x * blockDim.x + threadIdx.x;
  if (i >= n4) return;
  v4f v = ((const v4f*)in)[i];
  v4h o;
  o[0] = (_Float16)v[0]; o[1] = (_Float16)v[1];
  o[2] = (_Float16)v[2]; o[3] = (_Float16)v[3];
  ((v4h*)out)[i] = o;
}

// ---------------------------------------------------------------------------
// Tiled transpose + convert: in f32 [R x C] -> out f16 [C x R].
// Gives the GEMM B operand contiguous-K storage (b128 fragment loads).
// ---------------------------------------------------------------------------
__global__ void transpose_cvt_kernel(const float* __restrict__ in,
                                     _Float16* __restrict__ out, int R, int C) {
  __shared__ _Float16 tile[32][33];
  const int c0 = blockIdx.x * 32;
  const int r0 = blockIdx.y * 32;
  const int tx = threadIdx.x, ty = threadIdx.y;
#pragma unroll
  for (int j = 0; j < 4; ++j) {
    const int r = r0 + ty * 4 + j;
    tile[ty * 4 + j][tx] = (_Float16)in[r * C + c0 + tx];
  }
  __syncthreads();
#pragma unroll
  for (int j = 0; j < 4; ++j) {
    const int c = c0 + ty * 4 + j;
    out[c * R + r0 + tx] = tile[tx][ty * 4 + j];
  }
}

// ---------------------------------------------------------------------------
// Kernel 1: QKV GEMM.  A = xh f16 [4096x1024], B = Wqkv^T f16 [3072x1024].
// One wave -> 16x64 strip (4 WMMAs / k-step, A fragment reused 4x).
// Epilogue routes to q/k (head layout) and V transposed, + f32 k/v outputs.
// ---------------------------------------------------------------------------
__global__ void qkv_gemm_kernel(const _Float16* __restrict__ xh,
                                const _Float16* __restrict__ wt,   // [3072][1024]
                                const float* __restrict__ bqkv,
                                _Float16* __restrict__ qh,         // [B,H,T,DK] f16
                                _Float16* __restrict__ kh,         // [B,H,T,DK] f16
                                _Float16* __restrict__ vt,         // [B,H,DK,T] f16
                                float* __restrict__ k_out,
                                float* __restrict__ v_out) {
  const int K = EMBD_;
  const int nstrips = (3 * EMBD_) / 64;                 // 48
  const int tile = blockIdx.x * blockDim.y + threadIdx.y;
  const int mt = tile / nstrips;
  const int st = tile % nstrips;

  const int lane = threadIdx.x;
  const int half = lane >> 4;
  const int l16  = lane & 15;
  const int arow = mt * 16 + l16;

  v8f acc0 = {}, acc1 = {}, acc2 = {}, acc3 = {};
  const _Float16* arp = xh + arow * K;
  const _Float16* b0p = wt + (st * 64 +  0 + l16) * K;
  const _Float16* b1p = wt + (st * 64 + 16 + l16) * K;
  const _Float16* b2p = wt + (st * 64 + 32 + l16) * K;
  const _Float16* b3p = wt + (st * 64 + 48 + l16) * K;

  for (int k0 = 0; k0 < K; k0 += 32) {
    if (k0 + 32 < K) {
      __builtin_prefetch(arp + k0 + 32, 0, 1);
      __builtin_prefetch(b0p + k0 + 32, 0, 1);
    }
    const v8h* ap = (const v8h*)(arp + k0);
    const v16h a = cat16(ap[2 * half], ap[2 * half + 1]); // K = 16h+e contiguous

    const v8h* p0 = (const v8h*)(b0p + k0 + 16 * half); // K = 16h+e contiguous
    const v8h* p1 = (const v8h*)(b1p + k0 + 16 * half);
    const v8h* p2 = (const v8h*)(b2p + k0 + 16 * half);
    const v8h* p3 = (const v8h*)(b3p + k0 + 16 * half);
    acc0 = wmma_f16(a, cat16(p0[0], p0[1]), acc0);
    acc1 = wmma_f16(a, cat16(p1[0], p1[1]), acc1);
    acc2 = wmma_f16(a, cat16(p2[0], p2[1]), acc2);
    acc3 = wmma_f16(a, cat16(p3[0], p3[1]), acc3);
  }

  v8f accs[4] = {acc0, acc1, acc2, acc3};
#pragma unroll
  for (int i = 0; i < 4; ++i) {
    const int n   = st * 64 + i * 16 + l16;
    const int sec = n / EMBD_;
    const int c   = n % EMBD_;
    const int h   = c >> 6;
    const int d   = c & 63;
    const float bias = bqkv[n];
#pragma unroll
    for (int r = 0; r < 8; ++r) {
      const int m  = mt * 16 + r + 8 * half;
      const int bb = m / T_;
      const int t  = m % T_;
      const int bh = bb * NH_ + h;
      const float val = accs[i][r] + bias;
      const int idx = (bh * T_ + t) * DK_ + d;
      if (sec == 0) {
        qh[idx] = (_Float16)val;
      } else if (sec == 1) {
        k_out[idx] = val;
        kh[idx]    = (_Float16)val;
      } else {
        v_out[idx] = val;
        vt[(bh * DK_ + d) * T_ + t] = (_Float16)val;    // V transposed
      }
    }
  }
}

// ---------------------------------------------------------------------------
// Kernel 2: causal flash attention, one wave per (b,h, 16-query tile).
// All q/k/v traffic is f16 b128 loads; P transposed through LDS with an
// explicit CDNA5 s_wait_dscnt; output written f16 [B,T,C] for the projection.
// ---------------------------------------------------------------------------
__global__ void attn_kernel(const _Float16* __restrict__ qh,
                            const _Float16* __restrict__ kh,
                            const _Float16* __restrict__ vt,
                            _Float16* __restrict__ yh) {   // [B,T,C] f16
  __shared__ __align__(16) _Float16 lds_p[4][16 * 32];

  const int wave = threadIdx.y;
  const int qt   = blockIdx.x * 4 + wave;
  const int bh   = blockIdx.y;
  const int b    = bh / NH_;
  const int h    = bh % NH_;
  const int lane = threadIdx.x;
  const int half = lane >> 4;
  const int l16  = lane & 15;
  const int q0   = qt * 16;
  const float scale = 0.125f; // 1/sqrt(64)

  const _Float16* qb = qh + bh * (T_ * DK_);
  const _Float16* kb = kh + bh * (T_ * DK_);
  const _Float16* vb = vt + bh * (DK_ * T_);
  _Float16* pbuf = lds_p[wave];

  // Q A-fragments (d 0..31, 32..63), loaded once: 4x b128
  const v8h* qp = (const v8h*)(qb + (q0 + l16) * DK_);
  const v16h qa0 = cat16(qp[2 * half],     qp[2 * half + 1]);
  const v16h qa1 = cat16(qp[4 + 2 * half], qp[5 + 2 * half]);

  float m_run[8], l_run[8];
  v8f o0 = {}, o1 = {}, o2 = {}, o3 = {};
#pragma unroll
  for (int r = 0; r < 8; ++r) { m_run[r] = -1e30f; l_run[r] = 0.0f; }

  const int nsteps = (q0 + 16 + 31) / 32;
  for (int s = 0; s < nsteps; ++s) {
    const int k0 = s * 32;
    if (s + 1 < nsteps) __builtin_prefetch(kb + (k0 + 32 + l16) * DK_, 0, 1);

    // ---- S = Q K^T over 32 keys: each B fragment = 2x b128 (d contiguous)
    v8f S0 = {}, S1 = {};
    {
      const v8h* kp = (const v8h*)(kb + (k0 + l16) * DK_ + 16 * half);
      S0 = wmma_f16(qa0, cat16(kp[0], kp[1]), S0);   // d 0..31
      S0 = wmma_f16(qa1, cat16(kp[4], kp[5]), S0);   // d 32..63
    }
    {
      const v8h* kp = (const v8h*)(kb + (k0 + 16 + l16) * DK_ + 16 * half);
      S1 = wmma_f16(qa0, cat16(kp[0], kp[1]), S1);
      S1 = wmma_f16(qa1, cat16(kp[4], kp[5]), S1);
    }

    // ---- scale + causal mask + online softmax ----
#pragma unroll
    for (int r = 0; r < 8; ++r) {
      const int mq   = q0 + r + 8 * half;
      const int key0 = k0 + l16;
      const int key1 = k0 + 16 + l16;
      float v0 = S0[r] * scale;
      float v1 = S1[r] * scale;
      if (key0 > mq) v0 = -1e30f;
      if (key1 > mq) v1 = -1e30f;
      float mx = fmaxf(v0, v1);
      mx = fmaxf(mx, __shfl_xor(mx, 1, 16));
      mx = fmaxf(mx, __shfl_xor(mx, 2, 16));
      mx = fmaxf(mx, __shfl_xor(mx, 4, 16));
      mx = fmaxf(mx, __shfl_xor(mx, 8, 16));
      const float mnew  = fmaxf(m_run[r], mx);
      const float alpha = __expf(m_run[r] - mnew);
      const float p0 = __expf(v0 - mnew);
      const float p1 = __expf(v1 - mnew);
      float rs = p0 + p1;
      rs += __shfl_xor(rs, 1, 16);
      rs += __shfl_xor(rs, 2, 16);
      rs += __shfl_xor(rs, 4, 16);
      rs += __shfl_xor(rs, 8, 16);
      l_run[r] = l_run[r] * alpha + rs;
      m_run[r] = mnew;
      o0[r] *= alpha; o1[r] *= alpha; o2[r] *= alpha; o3[r] *= alpha;
      pbuf[(r + 8 * half) * 32 + l16]      = (_Float16)p0;
      pbuf[(r + 8 * half) * 32 + 16 + l16] = (_Float16)p1;
    }
    // single-wave LDS RAW: CDNA5 split DS counter + compiler fence
    asm volatile("s_wait_dscnt 0x0" ::: "memory");

    // ---- P A-fragment from LDS: 2x ds b128 ----
    const v8h* pp = (const v8h*)(pbuf + l16 * 32);
    const v16h pa = cat16(pp[2 * half], pp[2 * half + 1]);

    // ---- O += P V : V transposed -> contiguous keys, 2x b128 per chunk ----
    {
      const v8h* v0p = (const v8h*)(vb + ( 0 + l16) * T_ + k0 + 16 * half);
      const v8h* v1p = (const v8h*)(vb + (16 + l16) * T_ + k0 + 16 * half);
      const v8h* v2p = (const v8h*)(vb + (32 + l16) * T_ + k0 + 16 * half);
      const v8h* v3p = (const v8h*)(vb + (48 + l16) * T_ + k0 + 16 * half);
      o0 = wmma_f16(pa, cat16(v0p[0], v0p[1]), o0);
      o1 = wmma_f16(pa, cat16(v1p[0], v1p[1]), o1);
      o2 = wmma_f16(pa, cat16(v2p[0], v2p[1]), o2);
      o3 = wmma_f16(pa, cat16(v3p[0], v3p[1]), o3);
    }
  }

  // ---- epilogue: normalize, store f16 [B,T,C] (channel = h*64 + chunk*16 + n)
#pragma unroll
  for (int r = 0; r < 8; ++r) {
    const float inv = 1.0f / l_run[r];
    const int m = q0 + r + 8 * half;
    _Float16* dst = yh + (b * T_ + m) * EMBD_ + h * DK_;
    dst[l16]      = (_Float16)(o0[r] * inv);
    dst[16 + l16] = (_Float16)(o1[r] * inv);
    dst[32 + l16] = (_Float16)(o2[r] * inv);
    dst[48 + l16] = (_Float16)(o3[r] * inv);
  }
}

// ---------------------------------------------------------------------------
// Kernel 3: output projection. A = yh f16 [4096x1024], B = Wp^T f16.
// Same 16x64-per-wave register blocking as kernel 1.
// ---------------------------------------------------------------------------
__global__ void proj_gemm_kernel(const _Float16* __restrict__ yh,
                                 const _Float16* __restrict__ wt,  // [1024][1024]
                                 const float* __restrict__ bp,
                                 float* __restrict__ y) {
  const int K = EMBD_;
  const int N = EMBD_;
  const int nstrips = N / 64;                           // 16
  const int tile = blockIdx.x * blockDim.y + threadIdx.y;
  const int mt = tile / nstrips;
  const int st = tile % nstrips;

  const int lane = threadIdx.x;
  const int half = lane >> 4;
  const int l16  = lane & 15;
  const int arow = mt * 16 + l16;

  v8f acc0 = {}, acc1 = {}, acc2 = {}, acc3 = {};
  const _Float16* arp = yh + arow * K;
  const _Float16* b0p = wt + (st * 64 +  0 + l16) * K;
  const _Float16* b1p = wt + (st * 64 + 16 + l16) * K;
  const _Float16* b2p = wt + (st * 64 + 32 + l16) * K;
  const _Float16* b3p = wt + (st * 64 + 48 + l16) * K;

  for (int k0 = 0; k0 < K; k0 += 32) {
    if (k0 + 32 < K) {
      __builtin_prefetch(arp + k0 + 32, 0, 1);
      __builtin_prefetch(b0p + k0 + 32, 0, 1);
    }
    const v8h* ap = (const v8h*)(arp + k0);
    const v16h a = cat16(ap[2 * half], ap[2 * half + 1]);

    const v8h* p0 = (const v8h*)(b0p + k0 + 16 * half);
    const v8h* p1 = (const v8h*)(b1p + k0 + 16 * half);
    const v8h* p2 = (const v8h*)(b2p + k0 + 16 * half);
    const v8h* p3 = (const v8h*)(b3p + k0 + 16 * half);
    acc0 = wmma_f16(a, cat16(p0[0], p0[1]), acc0);
    acc1 = wmma_f16(a, cat16(p1[0], p1[1]), acc1);
    acc2 = wmma_f16(a, cat16(p2[0], p2[1]), acc2);
    acc3 = wmma_f16(a, cat16(p3[0], p3[1]), acc3);
  }

  v8f accs[4] = {acc0, acc1, acc2, acc3};
#pragma unroll
  for (int i = 0; i < 4; ++i) {
    const int n = st * 64 + i * 16 + l16;
    const float bias = bp[n];
#pragma unroll
    for (int r = 0; r < 8; ++r) {
      const int m = mt * 16 + r + 8 * half;
      y[m * N + n] = accs[i][r] + bias;
    }
  }
}

// ---------------------------------------------------------------------------
extern "C" void kernel_launch(void* const* d_in, const int* in_sizes, int n_in,
                              void* d_out, int out_size, void* d_ws, size_t ws_size,
                              hipStream_t stream) {
  const float* x    = (const float*)d_in[0];
  const float* Wqkv = (const float*)d_in[1];
  const float* bqkv = (const float*)d_in[2];
  const float* Wp   = (const float*)d_in[3];
  const float* bp   = (const float*)d_in[4];

  float* y     = (float*)d_out;                        // [B,T,C] f32
  float* k_out = y + (size_t)B_ * T_ * EMBD_;          // [B,H,T,DK] f32
  float* v_out = k_out + (size_t)B_ * T_ * EMBD_;      // [B,H,T,DK] f32

  // Workspace (f16), 40 MB total. xh aliased with yh (dead after QKV GEMM).
  _Float16* qh   = (_Float16*)d_ws;                          // 4M  (8 MB)
  _Float16* kh   = qh  + (size_t)B_ * T_ * EMBD_;            // 4M  (8 MB)
  _Float16* vt   = kh  + (size_t)B_ * T_ * EMBD_;            // 4M  (8 MB)
  _Float16* wqt  = vt  + (size_t)B_ * T_ * EMBD_;            // 3M  (6 MB)
  _Float16* wpt  = wqt + (size_t)3 * EMBD_ * EMBD_;          // 1M  (2 MB)
  _Float16* xh   = wpt + (size_t)EMBD_ * EMBD_;              // 4M  (8 MB)
  _Float16* yh16 = xh;                                       // alias

  // 0) converts / transposes (one-off, ~30 MB traffic)
  {
    const int n4 = (B_ * T_ * EMBD_) / 4;
    cvt_f16_kernel<<<n4 / 256, 256, 0, stream>>>(x, xh, n4);
    dim3 blk(32, 8);
    transpose_cvt_kernel<<<dim3((3 * EMBD_) / 32, EMBD_ / 32), blk, 0, stream>>>(
        Wqkv, wqt, EMBD_, 3 * EMBD_);
    transpose_cvt_kernel<<<dim3(EMBD_ / 32, EMBD_ / 32), blk, 0, stream>>>(
        Wp, wpt, EMBD_, EMBD_);
  }
  // 1) QKV GEMM: 256 row-tiles x 48 strips = 12288 waves, 8 waves/block
  {
    const int tiles = (B_ * T_ / 16) * ((3 * EMBD_) / 64);
    dim3 blk(32, 8);
    qkv_gemm_kernel<<<tiles / 8, blk, 0, stream>>>(xh, wqt, bqkv, qh, kh, vt,
                                                   k_out, v_out);
  }
  // 2) attention: grid (qtiles/4, B*H), 4 waves/block, 1 q-tile per wave
  {
    dim3 grd((T_ / 16) / 4, B_ * NH_);
    dim3 blk(32, 4);
    attn_kernel<<<grd, blk, 0, stream>>>(qh, kh, vt, yh16);
  }
  // 3) output projection: 256 x 16 strips = 4096 waves
  {
    const int tiles = (B_ * T_ / 16) * (EMBD_ / 64);
    dim3 blk(32, 8);
    proj_gemm_kernel<<<tiles / 8, blk, 0, stream>>>(yh16, wpt, bp, y);
  }
}
